// STGSAT_29102698398303
// MI455X (gfx1250) — matrix operand order repaired and checked by reading, not compiled, not run on hardware
//
#include <hip/hip_runtime.h>

// ---------------------------------------------------------------------------
// STGSAT on MI455X (gfx1250). All GEMMs via v_wmma_f32_16x16x32_f16 (wave32,
// f32 accumulate). Branch-free b128 staging + v_cvt_pk_rtz_f16_f32 packing.
// Gather-GEMM hoists edge indirection out of the K loop. Split-K for the
// skinny K=56512 GEMM. Irregular GAT softmax via float atomics.
// ---------------------------------------------------------------------------

typedef float    v8f  __attribute__((ext_vector_type(8)));
typedef _Float16 v16h __attribute__((ext_vector_type(16)));

constexpr int  BB   = 32;
constexpr int  LL   = 12;
constexpr int  NN   = 883;
constexpr int  HOR  = 12;
constexpr long R1   = (long)BB * NN;        // 28256
constexpr long R2   = (long)BB * LL;        // 384
constexpr int  E1   = 14128;
constexpr int  E2   = LL * LL;              // 144
constexpr long ET1  = (long)BB * E1;        // 452096
constexpr long ET2  = (long)BB * E2;        // 4608
constexpr long BIGF = R1 * 768;             // 21,700,608 floats

constexpr int BM = 64, BN = 64, BK = 32;
constexpr int LSTR  = 36;                   // LDS row stride in halves
constexpr int LSTRU = 18;                   // ... in dwords

__device__ __forceinline__ unsigned pkh(float a, float b) {
    typedef __fp16 h2v __attribute__((ext_vector_type(2)));
    union { h2v h; unsigned u; } u;
    u.h = __builtin_amdgcn_cvt_pkrtz(a, b);   // v_cvt_pk_rtz_f16_f32
    return u.u;
}

// ---------------- fragment helpers (ISA wave32 VGPR layouts) ---------------
__device__ __forceinline__ v16h load_afrag(const _Float16* Ash, int wm, int lane) {
    const int lr = lane & 15, hi = lane >> 4;
    const _Float16* arow = Ash + (wm * 16 + lr) * LSTR;
    const int ka0 = hi ? 8 : 0;        // lanes16-31 hold K 8..15 / 24..31
    v16h af;
    #pragma unroll
    for (int i = 0; i < 8; ++i) { af[i] = arow[ka0 + i]; af[8 + i] = arow[16 + ka0 + i]; }
    return af;
}
__device__ __forceinline__ v16h load_bfrag(const _Float16* Bth, int c0, int lane) {
    const int lr = lane & 15, hi = lane >> 4;
    const _Float16* bc = Bth + (c0 + lr) * LSTR;
    const int kb0 = hi ? 8 : 0;        // lanes16-31 hold K 8..15 / 24..31
    v16h bf;
    #pragma unroll
    for (int i = 0; i < 8; ++i) { bf[i] = bc[kb0 + i]; bf[8 + i] = bc[16 + kb0 + i]; }
    return bf;
}

__device__ __forceinline__ long perm_dst(int perm, int row, int col, int N, int pA, int pB) {
    switch (perm) {
      case 0: return (long)row * N + col;
      case 1: { const int b = row / pA, n = row % pA;           // pA=883,pB=12
                const int l = col >> 6, d = col & 63;
                return (((long)(b * pB + l)) * pA + n) * 64 + d; }
      case 2: { const int b = row / pA, l = row % pA;           // pA=12,pB=883
                const int n = col >> 6, d = col & 63;
                return (((long)(b * pB + n)) * pA + l) * 64 + d; }
      default:{ const int b = row / pA, n = row % pA;           // pA=883,pB=12
                return ((long)(b * pB + col)) * pA + n; }
    }
}

// ---------------------------------------------------------------------------
// Plain WMMA GEMM: C = act(A[M,K] @ W[K,N] + bias [+resid]); optional split-K
// (gridDim.z>1): partials atomically accumulated, bias added by z==0, C must
// be pre-zeroed, act/resid unused in that mode. Permuted scatter stores.
// ---------------------------------------------------------------------------
__global__ __launch_bounds__(256)
void gemm_wmma_plain(const float* __restrict__ A, const float* __restrict__ W,
                     const float* __restrict__ bias, const float* __restrict__ resid,
                     float* __restrict__ C, int M, int K, int N,
                     int act, int perm, int pA, int pB)
{
    __shared__ unsigned AsU[BM * LSTRU];
    __shared__ unsigned BtU[BN * LSTRU];
    _Float16* Ash = (_Float16*)AsU;
    _Float16* Bth = (_Float16*)BtU;

    const int tid = threadIdx.x, lane = tid & 31, wave = tid >> 5;
    const int wm = wave >> 1, wn = wave & 1;
    const int bm0 = blockIdx.y * BM, bn0 = blockIdx.x * BN;

    // staging maps (branch-free: clamp OOB rows/cols to a valid address;
    // the resulting garbage lanes are never stored)
    const int qa = tid & 7;                       // float4 slot in 32-k row
    const int ra = tid >> 3;                      // rows ra, ra+32
    const int colB = tid & 63, kpB = tid >> 6;    // B: col + k-pair base
    const long arow0 = (long)(bm0 + ra      < M ? bm0 + ra      : M - 1) * K;
    const long arow1 = (long)(bm0 + ra + 32 < M ? bm0 + ra + 32 : M - 1) * K;
    const int  gcolB = bn0 + colB < N ? bn0 + colB : N - 1;

    const int stepsTotal = K / BK;
    int s0 = 0, s1 = stepsTotal;
    const int zsplit = gridDim.z, z = blockIdx.z;
    if (zsplit > 1) {
        const int per = (stepsTotal + zsplit - 1) / zsplit;
        s0 = z * per; s1 = stepsTotal < s0 + per ? stepsTotal : s0 + per;
    }

    v8f acc[2] = {{}, {}};
    for (int s = s0; s < s1; ++s) {
        const int k0 = s * BK;
        const float4 va0 = *(const float4*)(A + arow0 + k0 + qa * 4);
        const float4 va1 = *(const float4*)(A + arow1 + k0 + qa * 4);
        float wb[8];
        #pragma unroll
        for (int kk = 0; kk < 4; ++kk) {
            const int k2 = kpB + kk * 4;
            wb[2*kk]   = W[(long)(k0 + 2*k2)     * N + gcolB];
            wb[2*kk+1] = W[(long)(k0 + 2*k2 + 1) * N + gcolB];
        }
        AsU[ra * LSTRU + qa*2]          = pkh(va0.x, va0.y);
        AsU[ra * LSTRU + qa*2 + 1]      = pkh(va0.z, va0.w);
        AsU[(ra+32) * LSTRU + qa*2]     = pkh(va1.x, va1.y);
        AsU[(ra+32) * LSTRU + qa*2 + 1] = pkh(va1.z, va1.w);
        #pragma unroll
        for (int kk = 0; kk < 4; ++kk)
            BtU[colB * LSTRU + kpB + kk*4] = pkh(wb[2*kk], wb[2*kk+1]);
        __syncthreads();

        const v16h af  = load_afrag(Ash, wm, lane);
        const v16h bf0 = load_bfrag(Bth, wn * 32,      lane);
        const v16h bf1 = load_bfrag(Bth, wn * 32 + 16, lane);
        acc[0] = __builtin_amdgcn_wmma_f32_16x16x32_f16(false, af, false, bf0,
                                                        (short)0, acc[0], false, false);
        acc[1] = __builtin_amdgcn_wmma_f32_16x16x32_f16(false, af, false, bf1,
                                                        (short)0, acc[1], false, false);
        __syncthreads();
    }

    const int lr = lane & 15, hi = lane >> 4;
    #pragma unroll
    for (int t = 0; t < 2; ++t) {
        const int col = bn0 + wn * 32 + t * 16 + lr;
        if (col >= N) continue;
        const float bv = bias ? bias[col] : 0.f;
        #pragma unroll
        for (int r = 0; r < 8; ++r) {
            const int row = bm0 + wm * 16 + hi * 8 + r;
            if (row >= M) continue;
            const long dst = perm_dst(perm, row, col, N, pA, pB);
            float v = acc[t][r];
            if (zsplit > 1) {
                if (z == 0) v += bv;
                atomicAdd(&C[dst], v);
            } else {
                v += bv;
                if (resid) v += resid[(long)row * N + col];
                if (act == 1) v = v > 0.f ? v : 0.f;
                C[dst] = v;
            }
        }
    }
}

// ---------------------------------------------------------------------------
// Gather WMMA GEMM for the GSAT extractor: A[row] = concat(emb[es],emb[et]),
// row-source offsets hoisted to LDS once per block; per-K-step uniform S/T
// select (K-steps never straddle the halfK boundary). Output row-major + ReLU.
// ---------------------------------------------------------------------------
__global__ __launch_bounds__(256)
void gemm_wmma_gather(const float* __restrict__ Asrc, const float* __restrict__ W,
                      const float* __restrict__ bias, float* __restrict__ C,
                      int M, int K, int N,
                      const int* __restrict__ ges, const int* __restrict__ get,
                      int E, int Nn, int halfK, int rowOffset)
{
    __shared__ unsigned AsU[BM * LSTRU];
    __shared__ unsigned BtU[BN * LSTRU];
    __shared__ unsigned offS[BM], offT[BM];
    _Float16* Ash = (_Float16*)AsU;
    _Float16* Bth = (_Float16*)BtU;

    const int tid = threadIdx.x, lane = tid & 31, wave = tid >> 5;
    const int wm = wave >> 1, wn = wave & 1;
    const int bm0 = blockIdx.y * BM, bn0 = blockIdx.x * BN;

    if (tid < BM) {                       // per-row gather offsets, K-invariant
        const int grow = bm0 + tid;
        unsigned oS = 0, oT = 0;
        if (grow < M) {
            const int rg = rowOffset + grow;
            const int b = rg / E, e = rg - b * E;
            oS = (unsigned)((b * Nn + ges[e]) * halfK);
            oT = (unsigned)((b * Nn + get[e]) * halfK);
        }
        offS[tid] = oS; offT[tid] = oT;
    }
    __syncthreads();

    const int qa = tid & 7, ra = tid >> 3;
    const int colB = tid & 63, kpB = tid >> 6;
    const int gcolB = bn0 + colB < N ? bn0 + colB : N - 1;

    v8f acc[2] = {{}, {}};
    const int stepsTotal = K / BK;
    for (int s = 0; s < stepsTotal; ++s) {
        const int k0  = s * BK;
        const unsigned* offp = (k0 < halfK) ? offS : offT;
        const int kk0 = k0 & (halfK - 1);
        const float4 va0 = *(const float4*)(Asrc + offp[ra]      + kk0 + qa * 4);
        const float4 va1 = *(const float4*)(Asrc + offp[ra + 32] + kk0 + qa * 4);
        float wb[8];
        #pragma unroll
        for (int kk = 0; kk < 4; ++kk) {
            const int k2 = kpB + kk * 4;
            wb[2*kk]   = W[(long)(k0 + 2*k2)     * N + gcolB];
            wb[2*kk+1] = W[(long)(k0 + 2*k2 + 1) * N + gcolB];
        }
        AsU[ra * LSTRU + qa*2]          = pkh(va0.x, va0.y);
        AsU[ra * LSTRU + qa*2 + 1]      = pkh(va0.z, va0.w);
        AsU[(ra+32) * LSTRU + qa*2]     = pkh(va1.x, va1.y);
        AsU[(ra+32) * LSTRU + qa*2 + 1] = pkh(va1.z, va1.w);
        #pragma unroll
        for (int kk = 0; kk < 4; ++kk)
            BtU[colB * LSTRU + kpB + kk*4] = pkh(wb[2*kk], wb[2*kk+1]);
        __syncthreads();

        const v16h af  = load_afrag(Ash, wm, lane);
        const v16h bf0 = load_bfrag(Bth, wn * 32,      lane);
        const v16h bf1 = load_bfrag(Bth, wn * 32 + 16, lane);
        acc[0] = __builtin_amdgcn_wmma_f32_16x16x32_f16(false, af, false, bf0,
                                                        (short)0, acc[0], false, false);
        acc[1] = __builtin_amdgcn_wmma_f32_16x16x32_f16(false, af, false, bf1,
                                                        (short)0, acc[1], false, false);
        __syncthreads();
    }

    const int lr = lane & 15, hi = lane >> 4;
    #pragma unroll
    for (int t = 0; t < 2; ++t) {
        const int col = bn0 + wn * 32 + t * 16 + lr;
        if (col >= N) continue;
        const float bv = bias[col];
        #pragma unroll
        for (int r = 0; r < 8; ++r) {
            const int row = bm0 + wm * 16 + hi * 8 + r;
            if (row >= M) continue;
            float v = acc[t][r] + bv;
            C[(long)row * N + col] = v > 0.f ? v : 0.f;   // ReLU
        }
    }
}

// ---------------------------------------------------------------------------
// helper VALU kernels
// ---------------------------------------------------------------------------
__device__ __forceinline__ unsigned f2ord(float f) {
    unsigned u = __float_as_uint(f);
    return (u & 0x80000000u) ? ~u : (u | 0x80000000u);
}
__device__ __forceinline__ float ord2f(unsigned u) {
    return (u & 0x80000000u) ? __uint_as_float(u & 0x7fffffffu)
                             : __uint_as_float(~u);
}

__global__ void zero_k(float* __restrict__ p, long total) {
    long t = (long)blockIdx.x * 256 + threadIdx.x;
    if (t < total) p[t] = 0.f;
}

__global__ void start_proj_k(const float* __restrict__ hist,
                             const float* __restrict__ sw,
                             const float* __restrict__ sb,
                             float* __restrict__ res, long total)
{
    long t = (long)blockIdx.x * 256 + threadIdx.x;
    if (t >= total) return;
    const int  col = (int)(t % 768);
    const long r   = t / 768;
    const int  b = (int)(r / NN), n = (int)(r % NN);
    const int  l = col >> 6, d = col & 63;
    const float* hp = hist + (((long)(b * LL + l) * NN + n) * 3);
    res[t] = hp[0] * sw[d] + hp[1] * sw[64 + d] + hp[2] * sw[128 + d] + sb[d];
}

__global__ void attn_scores_k(const float* __restrict__ h,
                              const float* __restrict__ asr,
                              const float* __restrict__ adt,
                              float* __restrict__ scs, float* __restrict__ scd,
                              int Rtot)
{
    int t = blockIdx.x * 256 + threadIdx.x;
    if (t >= Rtot * 4) return;
    const int r = t >> 2, hh = t & 3;
    const float* hp = h + (long)r * 128 + hh * 32;
    const float* ap = asr + hh * 32;
    const float* dp = adt + hh * 32;
    float s = 0.f, d = 0.f;
    #pragma unroll
    for (int i = 0; i < 32; ++i) { s += hp[i] * ap[i]; d += hp[i] * dp[i]; }
    scs[t] = s; scd[t] = d;
}

__global__ void seg_init_k(unsigned* __restrict__ segmax, float* __restrict__ segsum, int n)
{
    int t = blockIdx.x * 256 + threadIdx.x;
    if (t < n) { segmax[t] = 0x007FFFFFu /* f2ord(-inf) */; segsum[t] = 0.f; }
}

__global__ void edge_logits_k(const float* __restrict__ scs, const float* __restrict__ scd,
                              const int* __restrict__ es, const int* __restrict__ et,
                              int E, int Nn, float* __restrict__ ebuf,
                              unsigned* __restrict__ segmax, long total)
{
    long t = (long)blockIdx.x * 256 + threadIdx.x;
    if (t >= total) return;
    const long eg = t >> 2; const int hh = (int)(t & 3);
    const int b = (int)(eg / E), e = (int)(eg - (long)b * E);
    const int sdx = (b * Nn + es[e]) * 4 + hh;
    const int ddx = (b * Nn + et[e]) * 4 + hh;
    const float x  = scs[sdx] + scd[ddx];
    const float al = x > 0.f ? x : 0.2f * x;
    ebuf[t] = al;
    atomicMax(&segmax[ddx], f2ord(al));
}

__global__ void edge_exp_k(float* __restrict__ ebuf, const unsigned* __restrict__ segmax,
                           float* __restrict__ segsum, const int* __restrict__ et,
                           int E, int Nn, long total)
{
    long t = (long)blockIdx.x * 256 + threadIdx.x;
    if (t >= total) return;
    const long eg = t >> 2; const int hh = (int)(t & 3);
    const int b = (int)(eg / E), e = (int)(eg - (long)b * E);
    const int ddx = (b * Nn + et[e]) * 4 + hh;
    const float ex = __expf(ebuf[t] - ord2f(segmax[ddx]));
    ebuf[t] = ex;
    atomicAdd(&segsum[ddx], ex);
}

__global__ void edge_norm_k(float* __restrict__ ebuf, const float* __restrict__ segsum,
                            const int* __restrict__ et, int E, int Nn, long total)
{
    long t = (long)blockIdx.x * 256 + threadIdx.x;
    if (t >= total) return;
    const long eg = t >> 2; const int hh = (int)(t & 3);
    const int b = (int)(eg / E), e = (int)(eg - (long)b * E);
    ebuf[t] = ebuf[t] / segsum[(b * Nn + et[e]) * 4 + hh];
}

__global__ void agg_init_k(float* __restrict__ out, const float* __restrict__ bias, long total)
{
    long t = (long)blockIdx.x * 256 + threadIdx.x;
    if (t < total) out[t] = bias[t & 127];
}

__global__ void aggregate_k(float* __restrict__ out, const float* __restrict__ ebuf,
                            const float* __restrict__ att, const float* __restrict__ h,
                            const int* __restrict__ es, const int* __restrict__ et,
                            int E, int Nn, long total)
{
    long t = (long)blockIdx.x * 256 + threadIdx.x;
    if (t >= total) return;
    const int  c  = (int)(t & 127);
    const long eg = t >> 7;
    const int b = (int)(eg / E), e = (int)(eg - (long)b * E);
    const int hh = c >> 5;
    float w = ebuf[eg * 4 + hh];
    if (att) w *= att[eg];
    const float v = w * h[((long)(b * Nn + es[e])) * 128 + c];
    atomicAdd(&out[((long)(b * Nn + et[e])) * 128 + c], v);
}

__global__ void ext_final_k(const float* __restrict__ h2, const float* __restrict__ w3,
                            const float* __restrict__ b3, float* __restrict__ att,
                            int rows, int rowOffset)
{
    int t = blockIdx.x * 256 + threadIdx.x;
    if (t >= rows) return;
    const float* hp = h2 + (long)t * 128;
    float s = b3[0];
    #pragma unroll 8
    for (int i = 0; i < 128; ++i) s += hp[i] * w3[i];
    att[rowOffset + t] = 1.f / (1.f + __expf(-s));
}

__global__ void build_hid_k(const float* __restrict__ tsres, const float* __restrict__ tstemp,
                            const float* __restrict__ node_emb, const float* __restrict__ tid_emb,
                            const float* __restrict__ diw_emb, const float* __restrict__ hist,
                            float* __restrict__ hid, long total)
{
    long t = (long)blockIdx.x * 256 + threadIdx.x;
    if (t >= total) return;
    const int  c = (int)(t % 320);
    const long r = t / 320;
    const int b = (int)(r / NN), n = (int)(r % NN);
    float v;
    if (c < 64)       v = tsres[r * 64 + c];
    else if (c < 128) v = tstemp[r * 64 + (c - 64)];
    else if (c < 192) v = node_emb[n * 64 + (c - 128)];
    else if (c < 256) {
        const int tv = (int)hist[(((long)(b * LL + (LL - 1)) * NN + n) * 3) + 1];
        v = tid_emb[tv * 64 + (c - 192)];
    } else {
        const int dv = (int)hist[(((long)(b * LL + (LL - 1)) * NN + n) * 3) + 2];
        v = diw_emb[dv * 64 + (c - 256)];
    }
    hid[t] = v;
}

// ---------------------------------------------------------------------------
// host side
// ---------------------------------------------------------------------------
static inline int cdiv(long a, long b) { return (int)((a + b - 1) / b); }

static void gemm(hipStream_t st, const float* A, const float* W, const float* bias,
                 const float* resid, float* C, int M, int K, int N,
                 int act = 0, int perm = 0, int pA = 0, int pB = 0, int splitK = 1)
{
    dim3 grid(cdiv(N, BN), cdiv(M, BM), splitK);
    gemm_wmma_plain<<<grid, dim3(256), 0, st>>>(A, W, bias, resid, C, M, K, N,
                                                act, perm, pA, pB);
}

static void gemmg(hipStream_t st, const float* Asrc, const float* W, const float* bias,
                  float* C, int M, int K, int N,
                  const int* es, const int* et, int E, int Nn, int halfK, int ro)
{
    dim3 grid(cdiv(N, BN), cdiv(M, BM));
    gemm_wmma_gather<<<grid, dim3(256), 0, st>>>(Asrc, W, bias, C, M, K, N,
                                                 es, et, E, Nn, halfK, ro);
}

extern "C" void kernel_launch(void* const* d_in, const int* in_sizes, int n_in,
                              void* d_out, int out_size, void* d_ws, size_t ws_size,
                              hipStream_t stream)
{
    const float* hist      = (const float*)d_in[0];
    const int*   ei_sp     = (const int*)  d_in[1];
    const int*   ei_tp     = (const int*)  d_in[2];
    const float* node_emb  = (const float*)d_in[3];
    const float* tid_emb   = (const float*)d_in[4];
    const float* diw_emb   = (const float*)d_in[5];
    const float* start_w   = (const float*)d_in[6];
    const float* start_b   = (const float*)d_in[7];
    const float* ts_w      = (const float*)d_in[8];
    const float* ts_b      = (const float*)d_in[9];
    const float* tr_spat_w = (const float*)d_in[10];
    const float* tr_spat_b = (const float*)d_in[11];
    const float* inv_spat_w= (const float*)d_in[12];
    const float* inv_spat_b= (const float*)d_in[13];
    const float* tr_temp_w = (const float*)d_in[14];
    const float* tr_temp_b = (const float*)d_in[15];
    const float* inv_temp_w= (const float*)d_in[16];
    const float* inv_temp_b= (const float*)d_in[17];
    const float* gat_sp_w  = (const float*)d_in[18];
    const float* gat_sp_as = (const float*)d_in[19];
    const float* gat_sp_ad = (const float*)d_in[20];
    const float* gat_sp_b  = (const float*)d_in[21];
    const float* esp_w1    = (const float*)d_in[22];
    const float* esp_b1    = (const float*)d_in[23];
    const float* esp_w2    = (const float*)d_in[24];
    const float* esp_b2    = (const float*)d_in[25];
    const float* esp_w3    = (const float*)d_in[26];
    const float* esp_b3    = (const float*)d_in[27];
    const float* gat_tp_w  = (const float*)d_in[28];
    const float* gat_tp_as = (const float*)d_in[29];
    const float* gat_tp_ad = (const float*)d_in[30];
    const float* gat_tp_b  = (const float*)d_in[31];
    const float* etp_w1    = (const float*)d_in[32];
    const float* etp_b1    = (const float*)d_in[33];
    const float* etp_w2    = (const float*)d_in[34];
    const float* etp_b2    = (const float*)d_in[35];
    const float* etp_w3    = (const float*)d_in[36];
    const float* etp_b3    = (const float*)d_in[37];
    const float* enc_w1    = (const float*)d_in[38];
    const float* enc_b1    = (const float*)d_in[39];
    const float* enc_w2    = (const float*)d_in[40];
    const float* enc_b2    = (const float*)d_in[41];
    const float* reg_w     = (const float*)d_in[42];
    const float* reg_b     = (const float*)d_in[43];
    float* out = (float*)d_out;
    float* ws  = (float*)d_ws;

    const int* es1 = ei_sp;            const int* et1 = ei_sp + E1;
    const int* es2 = ei_tp;            const int* et2 = ei_tp + E2;

    // workspace layout (floats); slot0 time-shared:
    // res -> extractor h1/h2 -> spa_back_perm -> temp_back_perm
    float* slot0   = ws;
    float* h1c     = slot0;
    float* h2c     = slot0 + (long)28256 * 512;
    float* h1tp    = slot0;
    float* h2tp    = slot0 + (long)ET2 * 512;
    long   off     = BIGF;
    float* tsres   = ws + off; off += R1 * 64;
    float* tstemp  = ws + off; off += R1 * 64;
    float* spain   = ws + off; off += R1 * 128;
    float* hbuf    = ws + off; off += R1 * 128;
    float* emb     = ws + off; off += R1 * 128;
    float* spaout  = ws + off; off += R1 * 128;
    float* scs     = ws + off; off += R1 * 4;
    float* scd     = ws + off; off += R1 * 4;
    unsigned* segmax = (unsigned*)(ws + off); off += R1 * 4;
    float* segsum  = ws + off; off += R1 * 4;
    float* ebuf    = ws + off; off += ET1 * 4;
    float* attb    = ws + off; off += ET1;
    float* tempin  = ws + off; off += R2 * 128;
    float* htp     = ws + off; off += R2 * 128;
    float* embtp   = ws + off; off += R2 * 128;
    float* tpout   = ws + off; off += R2 * 128;
    float* scstp   = ws + off; off += R2 * 4;
    float* scdtp   = ws + off; off += R2 * 4;
    unsigned* segmaxtp = (unsigned*)(ws + off); off += R2 * 4;
    float* segsumtp= ws + off; off += R2 * 4;
    float* ebtp    = ws + off; off += ET2 * 4;
    float* atttp   = ws + off; off += ET2;
    float* hid     = ws + off; off += R1 * 320;
    float* enctmp  = ws + off; off += R1 * 320;
    (void)in_sizes; (void)n_in; (void)out_size; (void)ws_size;

    // 1) start projection directly into res = [B,N,L*DM]
    start_proj_k<<<cdiv(BIGF, 256), 256, 0, stream>>>(hist, start_w, start_b, slot0, BIGF);

    // 2) ts_res / spa_in
    gemm(stream, slot0, ts_w,      ts_b,      nullptr, tsres, (int)R1, 768, 64);
    gemm(stream, slot0, tr_spat_w, tr_spat_b, nullptr, spain, (int)R1, 768, 128);

    // 3) spatial GAT
    gemm(stream, spain, gat_sp_w, nullptr, nullptr, hbuf, (int)R1, 128, 128);
    attn_scores_k<<<cdiv(R1 * 4, 256), 256, 0, stream>>>(hbuf, gat_sp_as, gat_sp_ad, scs, scd, (int)R1);
    seg_init_k<<<cdiv(R1 * 4, 256), 256, 0, stream>>>(segmax, segsum, (int)(R1 * 4));
    edge_logits_k<<<cdiv(ET1 * 4, 256), 256, 0, stream>>>(scs, scd, es1, et1, E1, NN, ebuf, segmax, ET1 * 4);
    edge_exp_k   <<<cdiv(ET1 * 4, 256), 256, 0, stream>>>(ebuf, segmax, segsum, et1, E1, NN, ET1 * 4);
    edge_norm_k  <<<cdiv(ET1 * 4, 256), 256, 0, stream>>>(ebuf, segsum, et1, E1, NN, ET1 * 4);
    agg_init_k   <<<cdiv(R1 * 128, 256), 256, 0, stream>>>(emb, gat_sp_b, R1 * 128);
    aggregate_k  <<<cdiv(ET1 * 128, 256), 256, 0, stream>>>(emb, ebuf, nullptr, hbuf, es1, et1, E1, NN, ET1 * 128);

    // 4) GSAT extractor MLP over edges (chunked, gathered A)
    for (int c = 0; c < 16; ++c) {
        const int ro = c * 28256;
        gemmg(stream, emb, esp_w1, esp_b1, h1c, 28256, 256, 512, es1, et1, E1, NN, 128, ro);
        gemm (stream, h1c, esp_w2, esp_b2, nullptr, h2c, 28256, 512, 128, /*act*/1);
        ext_final_k<<<cdiv(28256, 256), 256, 0, stream>>>(h2c, esp_w3, esp_b3, attb, 28256, ro);
    }

    // 5) second GAT pass with edge mask
    agg_init_k <<<cdiv(R1 * 128, 256), 256, 0, stream>>>(spaout, gat_sp_b, R1 * 128);
    aggregate_k<<<cdiv(ET1 * 128, 256), 256, 0, stream>>>(spaout, ebuf, attb, hbuf, es1, et1, E1, NN, ET1 * 128);

    // 6) spa_back (permuted to [B,L,N*64]) then temporal transform (split-K)
    gemm(stream, spaout, inv_spat_w, inv_spat_b, nullptr, slot0, (int)R1, 128, 768,
         0, /*perm*/1, NN, LL);
    zero_k<<<cdiv(R2 * 128, 256), 256, 0, stream>>>(tempin, R2 * 128);
    gemm(stream, slot0, tr_temp_w, tr_temp_b, nullptr, tempin, (int)R2, 56512, 128,
         0, 0, 0, 0, /*splitK*/64);

    // 7) temporal GSAT
    gemm(stream, tempin, gat_tp_w, nullptr, nullptr, htp, (int)R2, 128, 128);
    attn_scores_k<<<cdiv(R2 * 4, 256), 256, 0, stream>>>(htp, gat_tp_as, gat_tp_ad, scstp, scdtp, (int)R2);
    seg_init_k<<<cdiv(R2 * 4, 256), 256, 0, stream>>>(segmaxtp, segsumtp, (int)(R2 * 4));
    edge_logits_k<<<cdiv(ET2 * 4, 256), 256, 0, stream>>>(scstp, scdtp, es2, et2, E2, LL, ebtp, segmaxtp, ET2 * 4);
    edge_exp_k   <<<cdiv(ET2 * 4, 256), 256, 0, stream>>>(ebtp, segmaxtp, segsumtp, et2, E2, LL, ET2 * 4);
    edge_norm_k  <<<cdiv(ET2 * 4, 256), 256, 0, stream>>>(ebtp, segsumtp, et2, E2, LL, ET2 * 4);
    agg_init_k   <<<cdiv(R2 * 128, 256), 256, 0, stream>>>(embtp, gat_tp_b, R2 * 128);
    aggregate_k  <<<cdiv(ET2 * 128, 256), 256, 0, stream>>>(embtp, ebtp, nullptr, htp, es2, et2, E2, LL, ET2 * 128);
    gemmg(stream, embtp, etp_w1, etp_b1, h1tp, (int)ET2, 256, 512, es2, et2, E2, LL, 128, 0);
    gemm (stream, h1tp, etp_w2, etp_b2, nullptr, h2tp, (int)ET2, 512, 128, /*act*/1);
    ext_final_k<<<cdiv(ET2, 256), 256, 0, stream>>>(h2tp, etp_w3, etp_b3, atttp, (int)ET2, 0);
    agg_init_k <<<cdiv(R2 * 128, 256), 256, 0, stream>>>(tpout, gat_tp_b, R2 * 128);
    aggregate_k<<<cdiv(ET2 * 128, 256), 256, 0, stream>>>(tpout, ebtp, atttp, htp, es2, et2, E2, LL, ET2 * 128);

    // 8) temp_back (permuted to [B*N, L*64]) then ts_temp
    gemm(stream, tpout, inv_temp_w, inv_temp_b, nullptr, slot0, (int)R2, 128, 56512,
         0, /*perm*/2, LL, NN);
    gemm(stream, slot0, ts_w, ts_b, nullptr, tstemp, (int)R1, 768, 64);

    // 9) concat -> hid
    build_hid_k<<<cdiv(R1 * 320, 256), 256, 0, stream>>>(tsres, tstemp, node_emb,
                                                         tid_emb, diw_emb, hist,
                                                         hid, R1 * 320);

    // 10) residual encoder x3
    for (int i = 0; i < 3; ++i) {
        gemm(stream, hid,    enc_w1 + (long)i * 320 * 320, enc_b1 + i * 320,
             nullptr, enctmp, (int)R1, 320, 320, /*act*/1);
        gemm(stream, enctmp, enc_w2 + (long)i * 320 * 320, enc_b2 + i * 320,
             hid, hid, (int)R1, 320, 320);
    }

    // 11) regression head -> d_out [B,HOR,N,1]
    gemm(stream, hid, reg_w, reg_b, nullptr, out, (int)R1, 320, HOR,
         0, /*perm*/3, NN, HOR);
}
